// SLinOSSMixer_38508676776664
// MI455X (gfx1250) — matrix operand order repaired and checked by reading, not compiled, not run on hardware
//
#include <hip/hip_runtime.h>

#define BTOT   2
#define TLEN   512
#define DMODEL 1024
#define DINNER 2048
#define NHEADS 32
#define DSTATE 32
#define DHEAD  64
#define KCONV  4
#define PN     (2*DINNER + 2*NHEADS)   /* 4160 */
#define BT     (BTOT*TLEN)             /* 1024 */
#define BCN    (NHEADS*4*DSTATE)       /* 4096 */

typedef __bf16 bf16_t;
typedef __attribute__((ext_vector_type(8)))  bf16_t bf16x8;
typedef __attribute__((ext_vector_type(16))) bf16_t v16bf;
typedef __attribute__((ext_vector_type(8)))  float  v8f;
typedef __attribute__((ext_vector_type(4)))  unsigned uint32x4;
typedef __attribute__((ext_vector_type(8)))  int      int32x8;
typedef __attribute__((ext_vector_type(4)))  int      int32x4;

#define USE_TDM __has_builtin(__builtin_amdgcn_tensor_load_to_lds)

__device__ __forceinline__ unsigned short f32_to_bf16(float f) {
  union { float f; unsigned u; } v; v.f = f;
  unsigned u = v.u;
  u += 0x7fffu + ((u >> 16) & 1u);           // round-to-nearest-even
  return (unsigned short)(u >> 16);
}

__global__ void cvt_f32_bf16_kernel(const float* __restrict__ in,
                                    unsigned short* __restrict__ out, int n) {
  int g = blockIdx.x * blockDim.x + threadIdx.x;
  if (g < n) out[g] = f32_to_bf16(in[g]);
}

#if USE_TDM
// Issue one TDM 2-D tile load: tile_w x tile_h elements (2 bytes each),
// row stride = row_stride elems, contiguous into LDS at lds_addr.
__device__ __forceinline__ void tdm_load_2d(unsigned lds_addr,
                                            const unsigned short* gptr,
                                            int tile_w, int tile_h,
                                            int row_stride, int tensor_w,
                                            int tensor_h) {
  unsigned long long ga = (unsigned long long)(uintptr_t)gptr;
  uint32x4 g0;
  g0[0] = 1u;                                       // count=1, user D#
  g0[1] = lds_addr;                                 // LDS byte address
  g0[2] = (unsigned)(ga & 0xffffffffu);             // global_addr[31:0]
  g0[3] = (unsigned)((ga >> 32) & 0x01ffffffu)      // global_addr[56:32]
          | (2u << 30);                             // type=2 (image)
  int32x8 g1;
  g1[0] = (1 << 16);                                // data_size=1 (2 bytes)
  g1[1] = (tensor_w & 0xffff) << 16;                // tensor_dim0[15:0]
  g1[2] = ((tensor_w >> 16) & 0xffff)               // tensor_dim0[31:16]
          | ((tensor_h & 0xffff) << 16);            // tensor_dim1[15:0]
  g1[3] = ((tensor_h >> 16) & 0xffff)               // tensor_dim1[31:16]
          | ((tile_w & 0xffff) << 16);              // tile_dim0
  g1[4] = (tile_h & 0xffff);                        // tile_dim1 (tile_dim2=0)
  g1[5] = row_stride;                               // tensor_dim0_stride[31:0]
  g1[6] = 0;
  g1[7] = 0;
  int32x4 z4 = {};
  int32x8 z8 = {};
  // 6-arg toolchain variant: (g0, g1, g2, g3, g4, cpol)
  __builtin_amdgcn_tensor_load_to_lds(g0, g1, z4, z4, z8, 0);
}
#endif

// C[m,n] = sum_k A[m,k]*B[n,k]; A:[M,K] bf16, B:[N,K] bf16, C:[M,N] f32.
// Block tile 128(M) x 64(N), 8 waves (4x2), wave tile 32x32 = 2x2 WMMA frags.
// K staged 32-wide through double-buffered LDS via the Tensor Data Mover.
// Requires M%128==0, N%64==0, K%32==0 (true for all three GEMMs here).
__global__ void __launch_bounds__(256)
gemm_bf16_wmma(const unsigned short* __restrict__ A,
               const unsigned short* __restrict__ Bm,
               float* __restrict__ C, int M, int N, int K) {
  __shared__ __align__(16) unsigned short lA[2][128 * 32];
  __shared__ __align__(16) unsigned short lB[2][64 * 32];
  const int tid  = threadIdx.x;
  const int lane = tid & 31;
  const int wv   = tid >> 5;
  const int wm   = wv & 3;      // 4 M-subtiles
  const int wn   = wv >> 2;     // 2 N-subtiles
  const int m0   = blockIdx.y * 128;
  const int n0   = blockIdx.x * 64;
  const int hf   = lane >> 4;   // half-wave select
  const int mr   = lane & 15;

  v8f acc[2][2];
#pragma unroll
  for (int i = 0; i < 2; ++i)
#pragma unroll
    for (int j = 0; j < 2; ++j) acc[i][j] = (v8f){};

#if USE_TDM
  const int steps = K >> 5;
  if (tid < 32) {   // wave 0 drives the TDM (EXEC-independent, per-wave issue)
    tdm_load_2d((unsigned)(uintptr_t)&lA[0][0], A + (size_t)m0 * K,
                32, 128, K, K, M);
    tdm_load_2d((unsigned)(uintptr_t)&lB[0][0], Bm + (size_t)n0 * K,
                32, 64, K, K, N);
  }
  for (int s = 0; s < steps; ++s) {
    const int cur = s & 1;
    if (tid < 32) __builtin_amdgcn_s_wait_tensorcnt((short)0);
    __syncthreads();                      // publish tile s to all waves
    if (tid < 32 && s + 1 < steps) {      // prefetch tile s+1 into other buffer
      const int k0 = (s + 1) << 5;
      tdm_load_2d((unsigned)(uintptr_t)&lA[cur ^ 1][0],
                  A + (size_t)m0 * K + k0, 32, 128, K, K, M);
      tdm_load_2d((unsigned)(uintptr_t)&lB[cur ^ 1][0],
                  Bm + (size_t)n0 * K + k0, 32, 64, K, K, N);
    }
    const bf16x8* pA = reinterpret_cast<const bf16x8*>(&lA[cur][0]);
    const bf16x8* pB = reinterpret_cast<const bf16x8*>(&lB[cur][0]);
#else
  for (int k0 = 0; k0 < K; k0 += 32) {
    __syncthreads();
#pragma unroll
    for (int i = 0; i < 2; ++i) {         // A tile: 512 x 16B chunks
      int c = i * 256 + tid;
      int row = c >> 2, kc = c & 3;
      const uint4* src = reinterpret_cast<const uint4*>(
          A + (size_t)(m0 + row) * K + k0 + kc * 8);
      *reinterpret_cast<uint4*>(&lA[0][row * 32 + kc * 8]) = *src;
    }
    {                                     // B tile: 256 chunks
      int row = tid >> 2, kc = tid & 3;
      const uint4* src = reinterpret_cast<const uint4*>(
          Bm + (size_t)(n0 + row) * K + k0 + kc * 8);
      *reinterpret_cast<uint4*>(&lB[0][row * 32 + kc * 8]) = *src;
    }
    if (k0 + 32 < K) {
      __builtin_prefetch(A + (size_t)(m0 + (tid >> 1)) * K + k0 + 32, 0, 0);
      __builtin_prefetch(Bm + (size_t)(n0 + (tid & 63)) * K + k0 + 32, 0, 0);
    }
    __syncthreads();
    const bf16x8* pA = reinterpret_cast<const bf16x8*>(&lA[0][0]);
    const bf16x8* pB = reinterpret_cast<const bf16x8*>(&lB[0][0]);
#endif

    // A frag (16x32 bf16): lane half hf holds K in [8hf,8hf+8) ++ [16+8hf,+8)
    v16bf afrag[2], bfrag[2];
#pragma unroll
    for (int mi = 0; mi < 2; ++mi) {
      int r = (wm * 32 + mi * 16 + mr) * 4;
      afrag[mi] = __builtin_shufflevector(pA[r + hf], pA[r + 2 + hf],
          0,1,2,3,4,5,6,7,8,9,10,11,12,13,14,15);
    }
    // B frag (32x16 bf16): lane half hf holds K in [8hf,8hf+8) ++ [16+8hf,+8)
#pragma unroll
    for (int ni = 0; ni < 2; ++ni) {
      int r = (wn * 32 + ni * 16 + mr) * 4;
      bfrag[ni] = __builtin_shufflevector(pB[r + hf], pB[r + 2 + hf],
          0,1,2,3,4,5,6,7,8,9,10,11,12,13,14,15);
    }
#pragma unroll
    for (int mi = 0; mi < 2; ++mi)
#pragma unroll
      for (int ni = 0; ni < 2; ++ni)
        acc[mi][ni] = __builtin_amdgcn_wmma_f32_16x16x32_bf16(
            false, afrag[mi], false, bfrag[ni], (short)0, acc[mi][ni],
            false, false);
  }

  // C layout: VGPR r -> M = 8*hf + r, N = lane%16
#pragma unroll
  for (int mi = 0; mi < 2; ++mi)
#pragma unroll
    for (int ni = 0; ni < 2; ++ni) {
      int row0 = m0 + wm * 32 + mi * 16 + hf * 8;
      int col  = n0 + wn * 32 + ni * 16 + mr;
#pragma unroll
      for (int r = 0; r < 8; ++r)
        C[(size_t)(row0 + r) * N + col] = acc[mi][ni][r];
    }
}

// depthwise conv (K=4, causal, per-batch zero pad) + SiLU; emit f32 + bf16
__global__ void conv_silu_kernel(const float* __restrict__ proj,
                                 const float* __restrict__ w,
                                 const float* __restrict__ b,
                                 float* __restrict__ value,
                                 unsigned short* __restrict__ value_bf) {
  int g = blockIdx.x * blockDim.x + threadIdx.x;
  if (g >= BT * DINNER) return;
  int bt = g / DINNER, c = g % DINNER;
  int t = bt % TLEN;
  float acc = b[c];
#pragma unroll
  for (int k = 0; k < KCONV; ++k) {
    int tt = t - (KCONV - 1) + k;
    if (tt >= 0)
      acc += proj[(size_t)(bt - (KCONV - 1) + k) * PN + DINNER + c] * w[c * KCONV + k];
  }
  float v = acc / (1.0f + __expf(-acc));
  value[g] = v;
  value_bf[g] = f32_to_bf16(v);
}

__global__ void params_kernel(const float* __restrict__ proj,
                              const float* __restrict__ dt_bias,
                              float* __restrict__ dt, float* __restrict__ kd) {
  int g = blockIdx.x * blockDim.x + threadIdx.x;
  if (g >= BT * NHEADS) return;
  int bt = g / NHEADS, h = g % NHEADS;
  float p0 = proj[(size_t)bt * PN + 2 * DINNER + 2 * h] + dt_bias[h];
  float p1 = proj[(size_t)bt * PN + 2 * DINNER + 2 * h + 1];
  float sp = (p0 > 20.f) ? p0 : log1pf(__expf(p0));
  dt[g] = fminf(fmaxf(sp, 1e-4f), 0.1f);
  kd[g] = 0.5f / (1.f + __expf(-p1));
}

// per-group (32 consecutive) L2 normalize: group == one wave (wave32)
__global__ void bc_norm_kernel(float* __restrict__ bc, int total) {
  int g = blockIdx.x * blockDim.x + threadIdx.x;
  if (g >= total) return;
  float v = bc[g];
  float ss = v * v;
#pragma unroll
  for (int off = 16; off >= 1; off >>= 1) ss += __shfl_xor(ss, off, 32);
  bc[g] = v / (sqrtf(ss) + 1e-8f);
}

// sequential scan over T; block = (b,h); lane = n (DSTATE=32), wave = 8-dh slab
__global__ void __launch_bounds__(256)
scan_kernel(const float* __restrict__ u, const float* __restrict__ dt,
            const float* __restrict__ kd, const float* __restrict__ bc,
            const float* __restrict__ A_log, float* __restrict__ y) {
  const int bh = blockIdx.x;
  const int b = bh / NHEADS, h = bh % NHEADS;
  const int lane = threadIdx.x & 31;   // n
  const int wv   = threadIdx.x >> 5;   // dh block (8 waves x 8 dh = 64)
  const float Aexp = __expf(A_log[h * DSTATE + lane]);
  float z[8], s[8];
#pragma unroll
  for (int j = 0; j < 8; ++j) { z[j] = 0.f; s[j] = 0.f; }

  for (int t = 0; t < TLEN; ++t) {
    const size_t bt = (size_t)b * TLEN + t;
    const float dtv = dt[bt * NHEADS + h];
    const float kv  = kd[bt * NHEADS + h];
    const float* bcp = bc + bt * BCN + h * (4 * DSTATE);
    const float Bz = bcp[lane];
    const float Bx = bcp[DSTATE + lane];
    const float Cz = bcp[2 * DSTATE + lane];
    const float Cx = bcp[3 * DSTATE + lane];
    const float S = 1.f / (1.f + dtv * kv + dtv * dtv * Aexp);
    const float* up = u + bt * DINNER + h * DHEAD + wv * 8;
    float p[8];
#pragma unroll
    for (int j = 0; j < 8; ++j) {
      float uu = up[j];
      z[j] = S * (z[j] - dtv * Aexp * s[j] + dtv * Bz * uu);
      s[j] = s[j] + dtv * z[j] + dtv * Bx * uu;
      p[j] = Cz * z[j] + Cx * s[j];
    }
#pragma unroll
    for (int j = 0; j < 8; ++j) {
#pragma unroll
      for (int off = 16; off >= 1; off >>= 1) p[j] += __shfl_xor(p[j], off, 32);
    }
    float* yp = y + bt * DINNER + h * DHEAD + wv * 8;
#pragma unroll
    for (int j = 0; j < 8; ++j)
      if (lane == j) yp[j] = p[j];
  }
}

__global__ void epilogue_kernel(const float* __restrict__ y,
                                const float* __restrict__ value,
                                const float* __restrict__ proj,
                                const float* __restrict__ skip,
                                unsigned short* __restrict__ y_bf) {
  int g = blockIdx.x * blockDim.x + threadIdx.x;
  if (g >= BT * DINNER) return;
  int bt = g / DINNER, c = g % DINNER;
  float gate = proj[(size_t)bt * PN + c];
  float v = (y[g] + skip[c] * value[g]) * (gate / (1.f + __expf(-gate)));
  y_bf[g] = f32_to_bf16(v);
}

extern "C" void kernel_launch(void* const* d_in, const int* in_sizes, int n_in,
                              void* d_out, int out_size, void* d_ws, size_t ws_size,
                              hipStream_t stream) {
  (void)in_sizes; (void)n_in; (void)out_size; (void)ws_size;
  const float* x         = (const float*)d_in[0];
  const float* W_in      = (const float*)d_in[1];
  const float* dw_weight = (const float*)d_in[2];
  const float* dw_bias   = (const float*)d_in[3];
  const float* W_bc      = (const float*)d_in[4];
  const float* W_out     = (const float*)d_in[5];
  const float* skip      = (const float*)d_in[6];
  const float* A_log     = (const float*)d_in[7];
  const float* dt_bias   = (const float*)d_in[8];

  char* ws = (char*)d_ws;
  size_t off = 0;
  auto alloc = [&](size_t bytes) -> void* {
    void* p = ws + off;
    off = (off + bytes + 255) & ~(size_t)255;
    return p;
  };
  unsigned short* x_bf     = (unsigned short*)alloc((size_t)BT * DMODEL * 2);
  unsigned short* win_bf   = (unsigned short*)alloc((size_t)PN * DMODEL * 2);
  unsigned short* wbc_bf   = (unsigned short*)alloc((size_t)BCN * DINNER * 2);
  unsigned short* wout_bf  = (unsigned short*)alloc((size_t)DMODEL * DINNER * 2);
  float*          proj     = (float*)alloc((size_t)BT * PN * 4);
  float*          value    = (float*)alloc((size_t)BT * DINNER * 4);
  unsigned short* value_bf = (unsigned short*)alloc((size_t)BT * DINNER * 2);
  float*          dtb      = (float*)alloc((size_t)BT * NHEADS * 4);
  float*          kdb      = (float*)alloc((size_t)BT * NHEADS * 4);
  float*          bcb      = (float*)alloc((size_t)BT * BCN * 4);
  float*          yb       = (float*)alloc((size_t)BT * DINNER * 4);
  unsigned short* y_bf     = (unsigned short*)alloc((size_t)BT * DINNER * 2);

  const int TPB = 256;
  auto blks = [](int n) { return (n + 255) / 256; };

  // fp32 -> bf16 conversions (activations + weights)
  cvt_f32_bf16_kernel<<<blks(BT * DMODEL), TPB, 0, stream>>>(x, x_bf, BT * DMODEL);
  cvt_f32_bf16_kernel<<<blks(PN * DMODEL), TPB, 0, stream>>>(W_in, win_bf, PN * DMODEL);
  cvt_f32_bf16_kernel<<<blks(BCN * DINNER), TPB, 0, stream>>>(W_bc, wbc_bf, BCN * DINNER);
  cvt_f32_bf16_kernel<<<blks(DMODEL * DINNER), TPB, 0, stream>>>(W_out, wout_bf, DMODEL * DINNER);

  // GEMM 1: proj[BT, 4160] = x[BT,1024] x W_in^T
  gemm_bf16_wmma<<<dim3(PN / 64, BT / 128), TPB, 0, stream>>>(
      x_bf, win_bf, proj, BT, PN, DMODEL);

  conv_silu_kernel<<<blks(BT * DINNER), TPB, 0, stream>>>(
      proj, dw_weight, dw_bias, value, value_bf);
  params_kernel<<<blks(BT * NHEADS), TPB, 0, stream>>>(proj, dt_bias, dtb, kdb);

  // GEMM 2: bc[BT, 4096] = value[BT,2048] x W_bc^T
  gemm_bf16_wmma<<<dim3(BCN / 64, BT / 128), TPB, 0, stream>>>(
      value_bf, wbc_bf, bcb, BT, BCN, DINNER);

  bc_norm_kernel<<<blks(BT * BCN), TPB, 0, stream>>>(bcb, BT * BCN);

  scan_kernel<<<BTOT * NHEADS, TPB, 0, stream>>>(value, dtb, kdb, bcb, A_log, yb);

  epilogue_kernel<<<blks(BT * DINNER), TPB, 0, stream>>>(yb, value, proj, skip, y_bf);

  // GEMM 3: out[BT, 1024] = y[BT,2048] x W_out^T
  gemm_bf16_wmma<<<dim3(DMODEL / 64, BT / 128), TPB, 0, stream>>>(
      y_bf, wout_bf, (float*)d_out, BT, DMODEL, DINNER);
}
